// DrugDiscoveryGNN_24988119728496
// MI455X (gfx1250) — matrix operand order repaired and checked by reading, not compiled, not run on hardware
//
#include <hip/hip_runtime.h>
#include <hip/hip_bf16.h>
#include <math.h>

// ---------------------------------------------------------------------------
// Types (plain ext-vector types: trivially-copyable, safe in unions)
// ---------------------------------------------------------------------------
typedef __attribute__((ext_vector_type(16))) _Float16 v16h;
typedef __attribute__((ext_vector_type(8)))  float    v8f;
typedef __attribute__((ext_vector_type(4)))  unsigned int v4u;
typedef __attribute__((ext_vector_type(4)))  float    v4f;

union FragH { v16h h; v4u u[2]; };

#define NNODES  100000
#define NEDGES  300000
#define INDIM   128
#define HIDDIM  256
#define NGRAPHS 2048

// ---------------------------------------------------------------------------
// Utility kernels
// ---------------------------------------------------------------------------
__global__ void zero_f32_kernel(float* __restrict__ p, int n) {
    for (int i = blockIdx.x * blockDim.x + threadIdx.x; i < n;
         i += gridDim.x * blockDim.x)
        p[i] = 0.0f;
}

__global__ void degree_kernel(const long long* __restrict__ dst,
                              float* __restrict__ deg, int E) {
    for (int e = blockIdx.x * blockDim.x + threadIdx.x; e < E;
         e += gridDim.x * blockDim.x)
        atomicAdd(&deg[(int)dst[e]], 1.0f);
}

__global__ void rsqrt_kernel(float* __restrict__ deg, int n) {
    for (int i = blockIdx.x * blockDim.x + threadIdx.x; i < n;
         i += gridDim.x * blockDim.x)
        deg[i] = rsqrtf(deg[i] + 1.0f);   // +1 for the self-loop
}

__global__ void cvt_f32_to_f16_kernel(const float* __restrict__ src,
                                      _Float16* __restrict__ dst, int n) {
    for (int i = blockIdx.x * blockDim.x + threadIdx.x; i < n;
         i += gridDim.x * blockDim.x)
        dst[i] = (_Float16)src[i];
}

// W is [K,N] row-major f32; Wt becomes [N,K] row-major f16 so that WMMA
// B-fragment reads are contiguous per lane.
__global__ void cvt_transpose_f16_kernel(const float* __restrict__ W,
                                         _Float16* __restrict__ Wt,
                                         int K, int N) {
    int total = K * N;
    for (int i = blockIdx.x * blockDim.x + threadIdx.x; i < total;
         i += gridDim.x * blockDim.x) {
        int n = i / K, k = i - n * K;
        Wt[i] = (_Float16)W[k * N + n];
    }
}

// ---------------------------------------------------------------------------
// WMMA GEMM:  C[M,N] = A[M,K] * B[K,N],  A f16 row-major, Bt = B^T f16 [N,K].
// Block = 128 threads = 4 waves. Each wave computes a 16x64 strip of C
// (4 independent 16x16 accumulators): the A fragment is loaded once per
// k-step and reused by 4 back-to-back independent v_wmma ops (no RAW chain,
// pipelines through the XDL and co-executes with address VALU).
// A/B fragments use contiguous k = 16*lg + j (same order in both); f32 C/D per CDNA5 ISA 7.12.2.
// ---------------------------------------------------------------------------
__global__ void wmma_gemm_kernel(const _Float16* __restrict__ A,
                                 const _Float16* __restrict__ Bt,
                                 float* __restrict__ C,
                                 int M, int K, int N) {
    const int lane = threadIdx.x & 31;
    const int wave = threadIdx.x >> 5;
    const int lr   = lane & 15;     // row (A) / col (B,D) within tile
    const int lg   = lane >> 4;     // lane-group select

    const int mbase  = blockIdx.x * 16;
    const int nstrip = wave * 64;                 // N == 256, 4 waves
    if (mbase >= M || nstrip >= N) return;        // uniform: EXEC stays all-1s

    const _Float16* Arow = A  + (size_t)(mbase + lr) * K;
    const _Float16* Brow = Bt + (size_t)(nstrip + lr) * K;   // tile 0 of strip
    const size_t tileStride = (size_t)16 * K;     // halves between B tiles

    v8f acc0 = {}, acc1 = {}, acc2 = {}, acc3 = {};
    for (int k0 = 0; k0 < K; k0 += 32) {
        FragH a, b0, b1, b2, b3;
        // A: j=0..15 -> K = k0 + lg*16 + j       (32B contiguous)
        a.u[0] = *reinterpret_cast<const v4u*>(Arow + k0 + lg * 16);
        a.u[1] = *reinterpret_cast<const v4u*>(Arow + k0 + lg * 16 + 8);
        // B: j=0..15 -> K = k0 + lg*16 + j       (32B contiguous per tile)
        const _Float16* bk = Brow + k0 + lg * 16;
        b0.u[0] = *reinterpret_cast<const v4u*>(bk);
        b0.u[1] = *reinterpret_cast<const v4u*>(bk + 8);
        b1.u[0] = *reinterpret_cast<const v4u*>(bk + tileStride);
        b1.u[1] = *reinterpret_cast<const v4u*>(bk + tileStride + 8);
        b2.u[0] = *reinterpret_cast<const v4u*>(bk + 2 * tileStride);
        b2.u[1] = *reinterpret_cast<const v4u*>(bk + 2 * tileStride + 8);
        b3.u[0] = *reinterpret_cast<const v4u*>(bk + 3 * tileStride);
        b3.u[1] = *reinterpret_cast<const v4u*>(bk + 3 * tileStride + 8);

        acc0 = __builtin_amdgcn_wmma_f32_16x16x32_f16(
            false, a.h, false, b0.h, (short)0, acc0, false, false);
        acc1 = __builtin_amdgcn_wmma_f32_16x16x32_f16(
            false, a.h, false, b1.h, (short)0, acc1, false, false);
        acc2 = __builtin_amdgcn_wmma_f32_16x16x32_f16(
            false, a.h, false, b2.h, (short)0, acc2, false, false);
        acc3 = __builtin_amdgcn_wmma_f32_16x16x32_f16(
            false, a.h, false, b3.h, (short)0, acc3, false, false);
    }

    // D: VGPR r, lanes 0-15 -> M=r ; lanes 16-31 -> M=r+8 ; N = lane&15
    v8f accs[4] = {acc0, acc1, acc2, acc3};
#pragma unroll
    for (int t = 0; t < 4; ++t) {
        const int n = nstrip + t * 16 + lr;
#pragma unroll
        for (int r = 0; r < 8; ++r) {
            int m = mbase + r + lg * 8;
            C[(size_t)m * N + n] = accs[t][r];
        }
    }
}

// ---------------------------------------------------------------------------
// Edge scatter: AGG[dst] += H[src] * dinv[src]*dinv[dst]
// One wave per edge; lane owns 8 consecutive features (2x float4 = 32B/lane).
// ---------------------------------------------------------------------------
__global__ void scatter_kernel(const long long* __restrict__ src,
                               const long long* __restrict__ dst,
                               const float* __restrict__ dinv,
                               const float* __restrict__ H,
                               float* __restrict__ AGG, int E) {
    const int lane  = threadIdx.x & 31;
    const int wave  = blockIdx.x * (blockDim.x >> 5) + (threadIdx.x >> 5);
    const int nwave = gridDim.x * (blockDim.x >> 5);

    for (int e = wave; e < E; e += nwave) {
        const int s = (int)src[e];
        const int d = (int)dst[e];
        const float w = dinv[s] * dinv[d];

        // prefetch next edge's source row (global_prefetch_b8)
        if (e + nwave < E) {
            int sn = (int)src[e + nwave];
            __builtin_prefetch(H + (size_t)sn * HIDDIM + (lane << 3), 0, 1);
        }

        const v4f* hp = reinterpret_cast<const v4f*>(
            H + (size_t)s * HIDDIM + (lane << 3));
        v4f h0 = hp[0];
        v4f h1 = hp[1];
        float* ap = AGG + (size_t)d * HIDDIM + (lane << 3);
        atomicAdd(ap + 0, h0.x * w);
        atomicAdd(ap + 1, h0.y * w);
        atomicAdd(ap + 2, h0.z * w);
        atomicAdd(ap + 3, h0.w * w);
        atomicAdd(ap + 4, h1.x * w);
        atomicAdd(ap + 5, h1.y * w);
        atomicAdd(ap + 6, h1.z * w);
        atomicAdd(ap + 7, h1.w * w);
    }
}

// H[i,f] = relu(AGG[i,f] + H[i,f]*dinv[i]^2 + b[f])   (in place)
__global__ void combine_relu_kernel(float* __restrict__ H,
                                    const float* __restrict__ AGG,
                                    const float* __restrict__ dinv,
                                    const float* __restrict__ bias,
                                    int total) {
    for (int i = blockIdx.x * blockDim.x + threadIdx.x; i < total;
         i += gridDim.x * blockDim.x) {
        int node = i >> 8;          // HIDDIM == 256
        int f    = i & 255;
        float di = dinv[node];
        float v  = AGG[i] + H[i] * di * di + bias[f];
        H[i] = fmaxf(v, 0.0f);
    }
}

// ---------------------------------------------------------------------------
// Pooling + head
// ---------------------------------------------------------------------------
__global__ void pool_sum_kernel(const float* __restrict__ ACT,
                                const long long* __restrict__ batch,
                                float* __restrict__ sums, int total) {
    for (int i = blockIdx.x * blockDim.x + threadIdx.x; i < total;
         i += gridDim.x * blockDim.x) {
        int node = i >> 8;
        int f    = i & 255;
        int g    = (int)batch[node];
        atomicAdd(&sums[(size_t)g * HIDDIM + f], ACT[i]);
    }
}

__global__ void pool_cnt_kernel(const long long* __restrict__ batch,
                                float* __restrict__ cnt, int n) {
    for (int i = blockIdx.x * blockDim.x + threadIdx.x; i < n;
         i += gridDim.x * blockDim.x)
        atomicAdd(&cnt[(int)batch[i]], 1.0f);
}

__global__ void head_kernel(const float* __restrict__ sums,
                            const float* __restrict__ cnt,
                            const float* __restrict__ Wf,
                            const float* __restrict__ bf,
                            float* __restrict__ out) {
    __shared__ float red[HIDDIM];
    const int g = blockIdx.x;
    const int t = threadIdx.x;
    float c = fmaxf(cnt[g], 1.0f);
    float p = sums[(size_t)g * HIDDIM + t] / c;
    red[t] = p * Wf[t];
    __syncthreads();
    for (int s = HIDDIM / 2; s > 0; s >>= 1) {
        if (t < s) red[t] += red[t + s];
        __syncthreads();
    }
    if (t == 0) {
        float z = red[0] + bf[0];
        out[g] = 1.0f / (1.0f + expf(-z));
    }
}

// ---------------------------------------------------------------------------
// Launch
// ---------------------------------------------------------------------------
extern "C" void kernel_launch(void* const* d_in, const int* in_sizes, int n_in,
                              void* d_out, int out_size, void* d_ws, size_t ws_size,
                              hipStream_t stream) {
    const float*     x     = (const float*)d_in[0];
    const long long* ei    = (const long long*)d_in[1];
    const long long* batch = (const long long*)d_in[2];
    const float*     W1    = (const float*)d_in[3];
    const float*     b1    = (const float*)d_in[4];
    const float*     W2    = (const float*)d_in[5];
    const float*     b2    = (const float*)d_in[6];
    const float*     Wf    = (const float*)d_in[7];
    const float*     bf    = (const float*)d_in[8];
    float*           out   = (float*)d_out;

    const long long* srcp = ei;            // edge_index[0]
    const long long* dstp = ei + NEDGES;   // edge_index[1]

    // -------- workspace carve-up (aligned 256B) --------
    char* ws = (char*)d_ws;
    size_t off = 0;
    auto take = [&](size_t bytes) -> char* {
        char* p = ws + off;
        off += (bytes + 255) & ~(size_t)255;
        return p;
    };
    float*    dinv = (float*)   take((size_t)NNODES * 4);
    _Float16* Xh   = (_Float16*)take((size_t)NNODES * INDIM * 2);
    _Float16* W1t  = (_Float16*)take((size_t)HIDDIM * INDIM * 2);   // [256,128]
    _Float16* W2t  = (_Float16*)take((size_t)HIDDIM * HIDDIM * 2);  // [256,256]
    float*    H    = (float*)   take((size_t)NNODES * HIDDIM * 4);
    float*    AGG  = (float*)   take((size_t)NNODES * HIDDIM * 4);
    _Float16* A1h  = (_Float16*)take((size_t)NNODES * HIDDIM * 2);
    float*    sums = (float*)   take((size_t)NGRAPHS * HIDDIM * 4);
    float*    cnt  = (float*)   take((size_t)NGRAPHS * 4);
    (void)ws_size;

    const int TB = 256;
    auto blocks = [&](int n) { int b = (n + TB - 1) / TB; return b > 65535 ? 65535 : b; };

    // 1) degrees -> dinv
    zero_f32_kernel<<<blocks(NNODES), TB, 0, stream>>>(dinv, NNODES);
    degree_kernel<<<blocks(NEDGES), TB, 0, stream>>>(dstp, dinv, NEDGES);
    rsqrt_kernel<<<blocks(NNODES), TB, 0, stream>>>(dinv, NNODES);

    // 2) f16 operands
    cvt_f32_to_f16_kernel<<<blocks(NNODES * INDIM), TB, 0, stream>>>(
        x, Xh, NNODES * INDIM);
    cvt_transpose_f16_kernel<<<blocks(INDIM * HIDDIM), TB, 0, stream>>>(
        W1, W1t, INDIM, HIDDIM);
    cvt_transpose_f16_kernel<<<blocks(HIDDIM * HIDDIM), TB, 0, stream>>>(
        W2, W2t, HIDDIM, HIDDIM);

    // 3) layer 1: H = Xh @ W1   (WMMA, one 16-row tile per block, 64 cols/wave)
    wmma_gemm_kernel<<<dim3(NNODES / 16), 128, 0, stream>>>(
        Xh, W1t, H, NNODES, INDIM, HIDDIM);
    zero_f32_kernel<<<blocks(NNODES * HIDDIM), TB, 0, stream>>>(AGG, NNODES * HIDDIM);
    scatter_kernel<<<2048, TB, 0, stream>>>(srcp, dstp, dinv, H, AGG, NEDGES);
    combine_relu_kernel<<<blocks(NNODES * HIDDIM), TB, 0, stream>>>(
        H, AGG, dinv, b1, NNODES * HIDDIM);

    // 4) layer 2: H = relu(GCN(ACT1))   (reuse H buffer for GEMM output)
    cvt_f32_to_f16_kernel<<<blocks(NNODES * HIDDIM), TB, 0, stream>>>(
        H, A1h, NNODES * HIDDIM);
    wmma_gemm_kernel<<<dim3(NNODES / 16), 128, 0, stream>>>(
        A1h, W2t, H, NNODES, HIDDIM, HIDDIM);
    zero_f32_kernel<<<blocks(NNODES * HIDDIM), TB, 0, stream>>>(AGG, NNODES * HIDDIM);
    scatter_kernel<<<2048, TB, 0, stream>>>(srcp, dstp, dinv, H, AGG, NEDGES);
    combine_relu_kernel<<<blocks(NNODES * HIDDIM), TB, 0, stream>>>(
        H, AGG, dinv, b2, NNODES * HIDDIM);

    // 5) global mean pool + linear head + sigmoid
    zero_f32_kernel<<<blocks(NGRAPHS * HIDDIM), TB, 0, stream>>>(sums, NGRAPHS * HIDDIM);
    zero_f32_kernel<<<blocks(NGRAPHS), TB, 0, stream>>>(cnt, NGRAPHS);
    pool_sum_kernel<<<blocks(NNODES * HIDDIM), TB, 0, stream>>>(
        H, batch, sums, NNODES * HIDDIM);
    pool_cnt_kernel<<<blocks(NNODES), TB, 0, stream>>>(batch, cnt, NNODES);
    head_kernel<<<NGRAPHS, HIDDIM, 0, stream>>>(sums, cnt, Wf, bf, out);
}
